// SAGPoolToAlphaZero_84834194030862
// MI455X (gfx1250) — matrix operand order repaired and checked by reading, not compiled, not run on hardware
//
#include <hip/hip_runtime.h>
#include <math.h>

typedef __attribute__((ext_vector_type(16))) _Float16 v16h;
typedef __attribute__((ext_vector_type(8)))  _Float16 v8h;
typedef __attribute__((ext_vector_type(8)))  float    v8f;

// ---------------------------------------------------------------- fill / convert
__global__ void fill_f32_kernel(float* __restrict__ p, float v, size_t n) {
  size_t i = (size_t)blockIdx.x * blockDim.x + threadIdx.x;
  if (i < n) p[i] = v;
}

__global__ void cvt_f16_kernel(const float* __restrict__ s, _Float16* __restrict__ d, size_t n) {
  size_t i = (size_t)blockIdx.x * blockDim.x + threadIdx.x;
  if (i < n) d[i] = (_Float16)s[i];
}

// Wt[n*K + k] = (f16) W[k*N + n]  (weights become column-major f16)
__global__ void transpose_f16_kernel(const float* __restrict__ w, _Float16* __restrict__ wt,
                                     int K, int N) {
  size_t i = (size_t)blockIdx.x * blockDim.x + threadIdx.x;
  if (i >= (size_t)K * N) return;
  int k = (int)(i / N);
  int n = (int)(i - (size_t)k * N);
  wt[(size_t)n * K + k] = (_Float16)w[i];
}

// ---------------------------------------------------------------- WMMA GEMM
// C[M,N] = A[M,K] @ W[K,N] (+bias)(+relu), A pre-converted f16 row-major,
// W pre-converted f16 column-major (Wt[N][K]). One wave computes a 16x64 strip
// (4 accumulators sharing one A fragment); 4 v_wmma per 32-wide k-step.
// Fragment layouts used here (wave32, contiguous k order, same for A and B):
//   A 16x32 f16 : lane -> row M=lane%16 ; elem e -> K = k0+(lane/16)*16+e
//                 (two contiguous 16B loads)
//   B 32x16 f16 : lane -> col N=lane%16 ; elem e -> K = k0+(lane/16)*16+e
//                 (16 contiguous halves of one Wt column = one 32B load)
//   C 16x16 f32 : vgpr r -> row M = r + (lane/16)*8 ; col N = lane%16
__global__ void gemm_wmma_kernel(const _Float16* __restrict__ A, const _Float16* __restrict__ Wt,
                                 const float* __restrict__ bias, float* __restrict__ C,
                                 int M, int K, int N, int relu)
{
  const int lane    = threadIdx.x & 31;
  const int wave    = threadIdx.x >> 5;
  const int strips  = N >> 6;                      // 64-column strips
  const int total   = (M >> 4) * strips;
  int tile = blockIdx.x * (blockDim.x >> 5) + wave;
  if (tile >= total) return;                       // wave-uniform: EXEC all-1s for WMMA
  const int tm = tile / strips, tn = tile - tm * strips;
  const int m0 = tm << 4, n0 = tn << 6;
  const int half = lane >> 4;
  const int l16  = lane & 15;

  const _Float16* __restrict__ Arow = A + (size_t)(m0 + l16) * K + half * 16;
  const _Float16* __restrict__ Bcol = Wt + (size_t)(n0 + l16) * K + half * 16;

  v8f a0 = {}, a1 = {}, a2 = {}, a3 = {};
  for (int k0 = 0; k0 < K; k0 += 32) {
    __builtin_prefetch(Arow + k0 + 128, 0, 1);     // global_prefetch_b8 on A stream
    v8h alo = *(const v8h*)(Arow + k0);
    v8h ahi = *(const v8h*)(Arow + k0 + 8);
    v16h af = __builtin_shufflevector(alo, ahi, 0, 1, 2, 3, 4, 5, 6, 7,
                                      8, 9, 10, 11, 12, 13, 14, 15);
    const _Float16* bb = Bcol + k0;
    v16h b0 = *(const v16h*)(bb);
    v16h b1 = *(const v16h*)(bb + (size_t)16 * K);
    v16h b2 = *(const v16h*)(bb + (size_t)32 * K);
    v16h b3 = *(const v16h*)(bb + (size_t)48 * K);
    a0 = __builtin_amdgcn_wmma_f32_16x16x32_f16(false, af, false, b0, (short)0, a0, false, false);
    a1 = __builtin_amdgcn_wmma_f32_16x16x32_f16(false, af, false, b1, (short)0, a1, false, false);
    a2 = __builtin_amdgcn_wmma_f32_16x16x32_f16(false, af, false, b2, (short)0, a2, false, false);
    a3 = __builtin_amdgcn_wmma_f32_16x16x32_f16(false, af, false, b3, (short)0, a3, false, false);
  }

  const int mrow = m0 + half * 8;
#pragma unroll
  for (int t = 0; t < 4; ++t) {
    const v8f acc = (t == 0) ? a0 : (t == 1) ? a1 : (t == 2) ? a2 : a3;
    const int col = n0 + t * 16 + l16;
    float bb = bias ? bias[col] : 0.0f;
#pragma unroll
    for (int r = 0; r < 8; ++r) {
      float v = acc[r] + bb;
      if (relu) v = fmaxf(v, 0.0f);
      C[(size_t)(mrow + r) * N + col] = v;
    }
  }
}

// ---------------------------------------------------------------- degree / coef
__global__ void deg_acc_kernel(const int* __restrict__ dst, const float* __restrict__ em,
                               float* __restrict__ deg, int n, int Ec, size_t total)
{
  size_t e = (size_t)blockIdx.x * blockDim.x + threadIdx.x;
  if (e >= total) return;
  int b = (int)(e / Ec);
  atomicAdd(&deg[(size_t)b * n + dst[e]], em[e]);
}

__global__ void coef_kernel(const int* __restrict__ src, const int* __restrict__ dst,
                            const float* __restrict__ em, const float* __restrict__ deg,
                            float* __restrict__ coef, int n, int Ec, size_t total)
{
  size_t e = (size_t)blockIdx.x * blockDim.x + threadIdx.x;
  if (e >= total) return;
  int b = (int)(e / Ec);
  coef[e] = rsqrtf(deg[(size_t)b * n + src[e]]) * rsqrtf(deg[(size_t)b * n + dst[e]]) * em[e];
}

// ---------------------------------------------------------------- edge scatter (features)
__global__ void feat_agg_kernel(const float* __restrict__ hW, const int* __restrict__ src,
                                const int* __restrict__ dst, const float* __restrict__ coef,
                                float* __restrict__ agg, int n, int F, int Ec)
{
  size_t e = blockIdx.x;               // one block per edge
  float c = coef[e];
  if (c == 0.0f) return;               // block-uniform: skip masked edges
  int b = (int)(e / Ec);
  const float* __restrict__ hs = hW + ((size_t)b * n + src[e]) * F;
  float* __restrict__ ad = agg + ((size_t)b * n + dst[e]) * F;
  for (int f = threadIdx.x; f < F; f += blockDim.x)
    atomicAdd(&ad[f], hs[f] * c);
}

__global__ void gcn_combine_kernel(float* __restrict__ agg, const float* __restrict__ hW,
                                   const float* __restrict__ deg, const float* __restrict__ bias,
                                   int F, int relu, size_t total)
{
  size_t i = (size_t)blockIdx.x * blockDim.x + threadIdx.x;
  if (i >= total) return;
  size_t node = i / F;
  int f = (int)(i - node * F);
  float v = agg[i] + hW[i] / deg[node] + bias[f];
  agg[i] = relu ? fmaxf(v, 0.0f) : v;
}

// ---------------------------------------------------------------- scorer (GCN with out-dim 1)
__global__ void matvec_kernel(const float* __restrict__ x, const float* __restrict__ w,
                              float* __restrict__ out, int F, size_t rows)
{
  size_t node = (size_t)blockIdx.x * (blockDim.x >> 5) + (threadIdx.x >> 5);
  if (node >= rows) return;
  int lane = threadIdx.x & 31;
  const float* xr = x + node * F;
  float s = 0.0f;
  for (int f = lane; f < F; f += 32) s += xr[f] * w[f];
  for (int off = 16; off > 0; off >>= 1) s += __shfl_down(s, off, 32);
  if (lane == 0) out[node] = s;
}

__global__ void scalar_agg_kernel(const float* __restrict__ sh, const int* __restrict__ src,
                                  const int* __restrict__ dst, const float* __restrict__ coef,
                                  float* __restrict__ sagg, int n, int Ec, size_t total)
{
  size_t e = (size_t)blockIdx.x * blockDim.x + threadIdx.x;
  if (e >= total) return;
  float c = coef[e];
  if (c == 0.0f) return;
  int b = (int)(e / Ec);
  atomicAdd(&sagg[(size_t)b * n + dst[e]], sh[(size_t)b * n + src[e]] * c);
}

__global__ void score_combine_kernel(float* __restrict__ sagg, const float* __restrict__ sh,
                                     const float* __restrict__ deg, const float* __restrict__ bg,
                                     size_t total)
{
  size_t i = (size_t)blockIdx.x * blockDim.x + threadIdx.x;
  if (i >= total) return;
  sagg[i] = sagg[i] + sh[i] / deg[i] + bg[0];
}

// ---------------------------------------------------------------- per-graph top-k (bitonic in LDS)
#define SORT_MAX 2048
__global__ void topk_pool_kernel(const float* __restrict__ score, int n, int k,
                                 int* __restrict__ perm, float* __restrict__ tval,
                                 int* __restrict__ newidx)
{
  __shared__ float skey[SORT_MAX];
  __shared__ int   sidx[SORT_MAX];
  const int b = blockIdx.x;
  const float* sc = score + (size_t)b * n;
  for (int i = threadIdx.x; i < n; i += blockDim.x) { skey[i] = sc[i]; sidx[i] = i; }
  __syncthreads();
  for (int ks = 2; ks <= n; ks <<= 1) {
    for (int j = ks >> 1; j > 0; j >>= 1) {
      for (int i = threadIdx.x; i < n; i += blockDim.x) {
        int ixj = i ^ j;
        if (ixj > i) {
          float ka = skey[i], kb = skey[ixj];
          int ia = sidx[i], ib = sidx[ixj];
          bool aFirst = (ka > kb) || (ka == kb && ia < ib);  // stable descending
          bool desc = ((i & ks) == 0);
          if (desc ? !aFirst : aFirst) {
            skey[i] = kb; skey[ixj] = ka; sidx[i] = ib; sidx[ixj] = ia;
          }
        }
      }
      __syncthreads();
    }
  }
  for (int i = threadIdx.x; i < n; i += blockDim.x) {
    int node = sidx[i];
    newidx[(size_t)b * n + node] = (i < k) ? i : -1;
    if (i < k) {
      perm[(size_t)b * k + i] = node;
      tval[(size_t)b * k + i] = tanhf(skey[i]);
    }
  }
}

__global__ void gather_pool_kernel(const float* __restrict__ x, const int* __restrict__ perm,
                                   const float* __restrict__ tval, float* __restrict__ xp,
                                   int n, int k, int F)
{
  size_t bi = blockIdx.x;               // b*k + i
  size_t b = bi / k;
  int node = perm[bi];
  float t = tval[bi];
  const float* xs = x + (b * n + node) * (size_t)F;
  float* xd = xp + bi * (size_t)F;
  for (int f = threadIdx.x; f < F; f += blockDim.x) xd[f] = xs[f] * t;
}

__global__ void edge_reindex_kernel(const int* __restrict__ src, const int* __restrict__ dst,
                                    const float* __restrict__ em, const int* __restrict__ newidx,
                                    int n, int* __restrict__ nsrc, int* __restrict__ ndst,
                                    float* __restrict__ nem, int Ec, size_t total)
{
  size_t e = (size_t)blockIdx.x * blockDim.x + threadIdx.x;
  if (e >= total) return;
  int b = (int)(e / Ec);
  int ns = newidx[(size_t)b * n + src[e]];
  int nd = newidx[(size_t)b * n + dst[e]];
  float keep = em[e] * (ns >= 0 ? 1.0f : 0.0f) * (nd >= 0 ? 1.0f : 0.0f);
  nsrc[e] = ns > 0 ? ns : 0;
  ndst[e] = nd > 0 ? nd : 0;
  nem[e] = keep;
}

// readout: z[b, f] += max_i xp[b,i,f] ; z[b, F+f] += mean_i xp[b,i,f]
__global__ void readout_kernel(const float* __restrict__ xp, float* __restrict__ z, int k, int F)
{
  size_t idx = (size_t)blockIdx.x * blockDim.x + threadIdx.x;  // b*F + f
  size_t b = idx / F;
  int f = (int)(idx - b * F);
  const float* p = xp + b * (size_t)k * F + f;
  float mx = -INFINITY, sm = 0.0f;
  for (int i = 0; i < k; ++i) { float v = p[(size_t)i * F]; mx = fmaxf(mx, v); sm += v; }
  z[b * (size_t)(2 * F) + f] += mx;
  z[b * (size_t)(2 * F) + F + f] += sm / (float)k;
}

// ---------------------------------------------------------------- heads
__global__ void log_softmax_kernel(const float* __restrict__ logits, float* __restrict__ out, int C)
{
  __shared__ float red[128];
  int b = blockIdx.x, t = threadIdx.x;   // blockDim == C == 128
  float v = logits[(size_t)b * C + t];
  red[t] = v; __syncthreads();
  for (int s = C >> 1; s > 0; s >>= 1) { if (t < s) red[t] = fmaxf(red[t], red[t + s]); __syncthreads(); }
  float mx = red[0]; __syncthreads();
  red[t] = expf(v - mx); __syncthreads();
  for (int s = C >> 1; s > 0; s >>= 1) { if (t < s) red[t] += red[t + s]; __syncthreads(); }
  float lse = logf(red[0]);
  out[(size_t)b * C + t] = v - mx - lse;
}

__global__ void vhead_kernel(const float* __restrict__ z2, const float* __restrict__ vW,
                             const float* __restrict__ vb, float* __restrict__ out, int F, int Brows)
{
  int b = blockIdx.x * blockDim.x + threadIdx.x;
  if (b >= Brows) return;
  float s = vb[0];
  for (int f = 0; f < F; ++f) s += z2[(size_t)b * F + f] * vW[f];
  out[b] = fmaxf(s, 0.0f);
}

// ================================================================ host
static inline dim3 gsz(size_t n, int t) { return dim3((unsigned)((n + (size_t)t - 1) / t)); }

extern "C" void kernel_launch(void* const* d_in, const int* in_sizes, int n_in,
                              void* d_out, int out_size, void* d_ws, size_t ws_size,
                              hipStream_t stream)
{
  (void)in_sizes; (void)n_in; (void)out_size; (void)ws_size;
  const int B = 64, N = 2048, E = 32768, F_IN = 256, NHID = 512, RULES = 128;
  const int K1 = 1024, K2 = 512, K3 = 256;

  const float* x    = (const float*)d_in[0];
  const int*   src0 = (const int*)d_in[1];
  const int*   dst0 = (const int*)d_in[2];
  const float* W1  = (const float*)d_in[3];  const float* b1  = (const float*)d_in[4];
  const float* Wp1 = (const float*)d_in[5];  const float* bp1 = (const float*)d_in[6];
  const float* W2  = (const float*)d_in[7];  const float* b2  = (const float*)d_in[8];
  const float* Wp2 = (const float*)d_in[9];  const float* bp2 = (const float*)d_in[10];
  const float* W3  = (const float*)d_in[11]; const float* b3  = (const float*)d_in[12];
  const float* Wp3 = (const float*)d_in[13]; const float* bp3 = (const float*)d_in[14];
  const float* l1W = (const float*)d_in[15]; const float* l1b = (const float*)d_in[16];
  const float* l2W = (const float*)d_in[17]; const float* l2b = (const float*)d_in[18];
  const float* piW = (const float*)d_in[19]; const float* pib = (const float*)d_in[20];
  const float* vW  = (const float*)d_in[21]; const float* vb  = (const float*)d_in[22];
  float* out = (float*)d_out;

  const size_t TE  = (size_t)B * E;            // edges total
  const size_t BIG = (size_t)B * N * NHID;     // largest node-feature buffer

  char* w = (char*)d_ws;
  auto alloc_f = [&](size_t nElem) { float*    p = (float*)w;    w += nElem * sizeof(float);    return p; };
  auto alloc_i = [&](size_t nElem) { int*      p = (int*)w;      w += nElem * sizeof(int);      return p; };
  auto alloc_h = [&](size_t nElem) { _Float16* p = (_Float16*)w; w += nElem * sizeof(_Float16); return p; };

  float*    bufA = alloc_f(BIG);
  float*    bufB = alloc_f(BIG);
  _Float16* hA16 = alloc_h((size_t)B * N * NHID / 2);   // max M*K among all GEMMs (33.5M halves)
  _Float16* wt16 = alloc_h((size_t)2 * NHID * NHID);    // max K*N = 1024*512
  float* em0  = alloc_f(TE);
  float* em1  = alloc_f(TE);
  float* em2  = alloc_f(TE);
  float* coef = alloc_f(TE);
  int*   s1   = alloc_i(TE);
  int*   d1   = alloc_i(TE);
  int*   s2   = alloc_i(TE);
  int*   d2   = alloc_i(TE);
  float* deg  = alloc_f((size_t)B * N);
  float* sh   = alloc_f((size_t)B * N);
  float* sagg = alloc_f((size_t)B * N);
  int*   nidx = alloc_i((size_t)B * N);
  int*   perm = alloc_i((size_t)B * K1);
  float* tval = alloc_f((size_t)B * K1);
  float* z    = alloc_f((size_t)B * 2 * NHID);
  float* z1   = alloc_f((size_t)B * NHID);
  float* z2   = alloc_f((size_t)B * (NHID / 2));
  float* lg   = alloc_f((size_t)B * RULES);

  auto fill = [&](float* p, float v, size_t n) {
    fill_f32_kernel<<<gsz(n, 256), dim3(256), 0, stream>>>(p, v, n);
  };
  auto gemm = [&](const float* A, const float* Wm, const float* bias, float* C,
                  int M, int K, int Nn, int relu) {
    cvt_f16_kernel<<<gsz((size_t)M * K, 256), dim3(256), 0, stream>>>(A, hA16, (size_t)M * K);
    transpose_f16_kernel<<<gsz((size_t)K * Nn, 256), dim3(256), 0, stream>>>(Wm, wt16, K, Nn);
    int tiles = (M / 16) * (Nn / 64);
    gemm_wmma_kernel<<<gsz((size_t)tiles, 8), dim3(256), 0, stream>>>(hA16, wt16, bias, C, M, K, Nn, relu);
  };

  // GCN layer: hW = xin@Wm ; deg/coef from edges ; agg = scatter(coef*hW[src]) ;
  // out(in aggbuf) = relu(agg + hW/deg + bias). deg/coef persist for the following pool.
  auto gcn_layer = [&](const float* xin, int n, int Fin, const int* sE, const int* dE,
                       const float* em, const float* Wm, const float* bias,
                       float* hWbuf, float* aggbuf) {
    int M = B * n;
    gemm(xin, Wm, nullptr, hWbuf, M, Fin, NHID, 0);
    fill(deg, 1.0f, (size_t)B * n);
    deg_acc_kernel<<<gsz(TE, 256), dim3(256), 0, stream>>>(dE, em, deg, n, E, TE);
    coef_kernel<<<gsz(TE, 256), dim3(256), 0, stream>>>(sE, dE, em, deg, coef, n, E, TE);
    fill(aggbuf, 0.0f, (size_t)M * NHID);
    feat_agg_kernel<<<dim3((unsigned)TE), dim3(256), 0, stream>>>(hWbuf, sE, dE, coef, aggbuf, n, NHID, E);
    gcn_combine_kernel<<<gsz((size_t)M * NHID, 256), dim3(256), 0, stream>>>(
        aggbuf, hWbuf, deg, bias, NHID, 1, (size_t)M * NHID);
  };

  // SAGPool: scorer GCN (reuses deg/coef), topk, gather+tanh scale, edge reindex, readout.
  auto pool = [&](const float* xin, int n, int k, const int* sE, const int* dE,
                  const float* em, const float* Wg, const float* bg, float* xpout,
                  int* nsrc, int* ndst, float* nem) {
    size_t rows = (size_t)B * n;
    matvec_kernel<<<gsz(rows, 8), dim3(256), 0, stream>>>(xin, Wg, sh, NHID, rows);
    fill(sagg, 0.0f, rows);
    scalar_agg_kernel<<<gsz(TE, 256), dim3(256), 0, stream>>>(sh, sE, dE, coef, sagg, n, E, TE);
    score_combine_kernel<<<gsz(rows, 256), dim3(256), 0, stream>>>(sagg, sh, deg, bg, rows);
    topk_pool_kernel<<<dim3(B), dim3(256), 0, stream>>>(sagg, n, k, perm, tval, nidx);
    gather_pool_kernel<<<dim3((unsigned)(B * k)), dim3(256), 0, stream>>>(xin, perm, tval, xpout, n, k, NHID);
    if (nsrc)
      edge_reindex_kernel<<<gsz(TE, 256), dim3(256), 0, stream>>>(sE, dE, em, nidx, n, nsrc, ndst, nem, E, TE);
    readout_kernel<<<gsz((size_t)B * NHID, 256), dim3(256), 0, stream>>>(xpout, z, k, NHID);
  };

  // ---------- pipeline ----------
  fill(em0, 1.0f, TE);
  fill(z, 0.0f, (size_t)B * 2 * NHID);

  // layer 1: h1 -> bufB ; hp1 -> bufA
  gcn_layer(x, N, F_IN, src0, dst0, em0, W1, b1, bufA, bufB);
  pool(bufB, N, K1, src0, dst0, em0, Wp1, bp1, bufA, s1, d1, em1);

  // layer 2: h2 -> bufA ; hp2 -> bufB
  gcn_layer(bufA, K1, NHID, s1, d1, em1, W2, b2, bufB, bufA);
  pool(bufA, K1, K2, s1, d1, em1, Wp2, bp2, bufB, s2, d2, em2);

  // layer 3: h3 -> bufB ; hp3 -> bufA (no edge remap needed after last pool)
  gcn_layer(bufB, K2, NHID, s2, d2, em2, W3, b3, bufA, bufB);
  pool(bufB, K2, K3, s2, d2, em2, Wp3, bp3, bufA, nullptr, nullptr, nullptr);

  // ---------- MLP head (WMMA GEMMs) ----------
  gemm(z,  l1W, l1b, z1, B, 2 * NHID, NHID, 1);
  gemm(z1, l2W, l2b, z2, B, NHID, NHID / 2, 1);
  gemm(z2, piW, pib, lg, B, NHID / 2, RULES, 0);
  log_softmax_kernel<<<dim3(B), dim3(RULES), 0, stream>>>(lg, out, RULES);
  vhead_kernel<<<dim3(1), dim3(64), 0, stream>>>(z2, vW, vb, out + (size_t)B * RULES, NHID / 2, B);
}
